// BD3LM_30245159699071
// MI455X (gfx1250) — matrix operand order repaired and checked by reading, not compiled, not run on hardware
//
#include <hip/hip_runtime.h>
#include <hip/hip_bf16.h>
#include <math.h>

// ---------------------------------------------------------------------------
// BD3LM forward on MI455X (gfx1250): bf16 WMMA GEMMs, async-LDS K staging,
// TDM (tensor_load_to_lds) V staging, flash-style block-causal attention.
// Model: V=32000 D=768 H=12 L=12 DF=3072 B=2 S=1024 BS=16 HD=64, N=B*S=2048.
// Output: d_out[0] = loss, d_out[1..] = logits (B*S*V f32).
// ---------------------------------------------------------------------------

#define CV 32000
#define CD 768
#define CH 12
#define CL 12
#define CDF 3072
#define CB 2
#define CS 1024
#define CN (CB * CS)      // 2048 token rows
#define CHD 64
#define MASK_ID (CV - 1)

typedef __attribute__((ext_vector_type(16))) __bf16    v16bf;
typedef __attribute__((ext_vector_type(8)))  float     v8f;
typedef __attribute__((ext_vector_type(4)))  unsigned  v4u;
typedef __attribute__((ext_vector_type(8)))  int       v8i;
typedef __attribute__((ext_vector_type(4)))  int       v4i;

#if __has_builtin(__builtin_amdgcn_tensor_load_to_lds)
#define HAVE_TDM 1
#else
#define HAVE_TDM 0
#endif

// ---------------- WMMA helpers (k = 16*half + elem in both A and B) ---

__device__ __forceinline__ v8f wmma_bf16(v16bf a, v16bf b, v8f c) {
  // v_wmma_f32_16x16x32_bf16: D = A(16x32) * B(32x16) + C(16x16 f32)
  return __builtin_amdgcn_wmma_f32_16x16x32_bf16(
      /*neg_a=*/false, a, /*neg_b=*/false, b,
      /*c_mod=*/(short)0, c, /*reuse_a=*/false, /*reuse_b=*/false);
}

// A-matrix 16x32 bf16: lane m (0..15) row M=m; half = lane>>4 selects K groups
// {16h+0..7} in elems 0..7 and {16h+8..15} in elems 8..15.
__device__ __forceinline__ v16bf loadA_f32(const float* __restrict__ row,
                                           int kbase, int half) {
  v16bf a;
  const float* p0 = row + kbase + 16 * half;
  const float* p1 = p0 + 8;
#pragma unroll
  for (int j = 0; j < 8; ++j) a[j] = (__bf16)p0[j];
#pragma unroll
  for (int j = 0; j < 8; ++j) a[8 + j] = (__bf16)p1[j];
  return a;
}

// B-matrix 32x16 bf16 (column per lane): lane n holds column n,
// K = kbase + 16*half + j for elems j=0..15. Source row is W[n][*] (K-major),
// since B = W^T.
__device__ __forceinline__ v16bf loadB_f32(const float* __restrict__ row,
                                           int kbase) {
  v16bf b;
#pragma unroll
  for (int j = 0; j < 16; ++j) b[j] = (__bf16)row[kbase + j];
  return b;
}

// ---------------- TDM: 2D tile load Global -> LDS (D# per §8 of 08_async) ---
#if HAVE_TDM
__device__ __forceinline__ void tdm_load_2d(unsigned ldsoff, const void* gaddr,
                                            unsigned rows, unsigned rowlen,
                                            unsigned long long stride_elems) {
  const unsigned long long ga = (unsigned long long)(size_t)gaddr;
  v4u g0;
  g0.x = 1u;                                   // count=1, user mode
  g0.y = ldsoff;                               // LDS byte address
  g0.z = (unsigned)(ga & 0xFFFFFFFFu);         // global_addr[31:0]
  g0.w = (unsigned)((ga >> 32) & 0x1FFFFFFu)   // global_addr[56:32]
         | (2u << 30);                         // type = 2 ("image")
  v8i g1;
  g1[0] = (int)(2u << 16);                     // data_size = 4 bytes
  g1[1] = (int)((rowlen & 0xFFFFu) << 16);     // tensor_dim0[15:0]
  g1[2] = (int)(((rowlen >> 16) & 0xFFFFu) |   // tensor_dim0[31:16]
                ((rows & 0xFFFFu) << 16));     // tensor_dim1[15:0]
  g1[3] = (int)(((rows >> 16) & 0xFFFFu) |     // tensor_dim1[31:16]
                ((rowlen & 0xFFFFu) << 16));   // tile_dim0
  g1[4] = (int)(rows & 0xFFFFu);               // tile_dim1 (tile_dim2 = 0)
  g1[5] = (int)(unsigned)(stride_elems & 0xFFFFFFFFull);  // dim0_stride lo32
  g1[6] = (int)(unsigned)((stride_elems >> 32) & 0xFFFFull);
  g1[7] = 0;
  const v4i z4 = {0, 0, 0, 0};
  const v8i z8 = {0, 0, 0, 0, 0, 0, 0, 0};
  __builtin_amdgcn_tensor_load_to_lds(g0, g1, z4, z4, z8, 0);
}
#endif

// ---------------- GEMM: C[M,N] = A[M,K] @ W[N,K]^T + bias (+resid) ----------
// 8 waves / block in a 2(m) x 4(n) grid; each wave owns a 32x64 C macro-tile
// (2x4 WMMA tiles): 2 A-loads + 4 B-loads feed 8 wmma per k-step.
// Block tile: 64 x 256. M % 64 == 0, N % 256 == 0, K % 32 == 0 for all calls.
template <bool HAS_RESID>
__global__ __launch_bounds__(256) void gemm_bf16_wmma(
    const float* __restrict__ A, const float* __restrict__ W,
    const float* __restrict__ bias, const float* __restrict__ resid,
    float* __restrict__ C, int M, int N, int K) {
  const int wave = threadIdx.x >> 5;
  const int lane = threadIdx.x & 31;
  const int half = lane >> 4, r = lane & 15;
  const int wm = wave >> 2;                    // 0..1
  const int wn = wave & 3;                     // 0..3
  const int m0 = blockIdx.y * 64 + wm * 32;    // wave M base (2 tiles)
  const int n0 = blockIdx.x * 256 + wn * 64;   // wave N base (4 tiles)

  const float* Arow0 = A + (size_t)(m0 + r) * K;
  const float* Arow1 = Arow0 + (size_t)16 * K;
  const float* Wr0 = W + (size_t)(n0 + r) * K;
  const float* Wr1 = Wr0 + (size_t)16 * K;
  const float* Wr2 = Wr0 + (size_t)32 * K;
  const float* Wr3 = Wr0 + (size_t)48 * K;

  const v8f vz = {0.f, 0.f, 0.f, 0.f, 0.f, 0.f, 0.f, 0.f};
  v8f c0[4] = {vz, vz, vz, vz};
  v8f c1[4] = {vz, vz, vz, vz};

  for (int k0 = 0; k0 < K; k0 += 32) {
    if (k0 + 64 < K) {                         // -> global_prefetch_b8
      __builtin_prefetch(Wr0 + k0 + 64, 0, 3);
      __builtin_prefetch(Wr1 + k0 + 64, 0, 3);
      __builtin_prefetch(Wr2 + k0 + 64, 0, 3);
      __builtin_prefetch(Wr3 + k0 + 64, 0, 3);
    }
    const v16bf a0 = loadA_f32(Arow0, k0, half);
    const v16bf a1 = loadA_f32(Arow1, k0, half);
    const int kb = k0 + 16 * half;
    const v16bf b0 = loadB_f32(Wr0, kb);
    const v16bf b1 = loadB_f32(Wr1, kb);
    const v16bf b2 = loadB_f32(Wr2, kb);
    const v16bf b3 = loadB_f32(Wr3, kb);
    c0[0] = wmma_bf16(a0, b0, c0[0]);
    c0[1] = wmma_bf16(a0, b1, c0[1]);
    c0[2] = wmma_bf16(a0, b2, c0[2]);
    c0[3] = wmma_bf16(a0, b3, c0[3]);
    c1[0] = wmma_bf16(a1, b0, c1[0]);
    c1[1] = wmma_bf16(a1, b1, c1[1]);
    c1[2] = wmma_bf16(a1, b2, c1[2]);
    c1[3] = wmma_bf16(a1, b3, c1[3]);
  }

#pragma unroll
  for (int t = 0; t < 4; ++t) {
    const int col = n0 + 16 * t + r;
    const float bv = bias[col];
#pragma unroll
    for (int vr = 0; vr < 8; ++vr) {
      const int row0 = m0 + vr + 8 * half;
      const int row1 = row0 + 16;
      const size_t i0 = (size_t)row0 * N + col;
      const size_t i1 = (size_t)row1 * N + col;
      float v0 = c0[t][vr] + bv;
      float v1 = c1[t][vr] + bv;
      if (HAS_RESID) { v0 += resid[i0]; v1 += resid[i1]; }
      C[i0] = v0;
      C[i1] = v1;
    }
  }
}

// ---------------- Embedding gather + rotary (first D/2 channels) ------------
__global__ __launch_bounds__(256) void embed_rotary(
    const int* __restrict__ ids, const float* __restrict__ emb,
    float* __restrict__ x) {
  const int tok = blockIdx.x;        // 0..N-1
  const int s   = tok % CS;
  const float* e = emb + (size_t)ids[tok] * CD;
  float* xr = x + (size_t)tok * CD;
  for (int p = threadIdx.x; p < 192; p += blockDim.x) {
    const float fr  = __powf(10000.f, -(float)(2 * p) / 384.f);
    const float ang = (float)s * fr;
    const float cs = __cosf(ang), sn = __sinf(ang);
    const float x0 = e[2 * p], x1 = e[2 * p + 1];
    xr[2 * p]     = x0 * cs - x1 * sn;
    xr[2 * p + 1] = x0 * sn + x1 * cs;
  }
  for (int d = 384 + threadIdx.x; d < CD; d += blockDim.x) xr[d] = e[d];
}

// ---------------- Block reductions (256 threads = 8 waves) ------------------
__device__ __forceinline__ float blockReduceSum256(float v, float* sh) {
  const int lane = threadIdx.x & 31, wid = threadIdx.x >> 5;
#pragma unroll
  for (int off = 16; off; off >>= 1) v += __shfl_xor(v, off, 32);
  __syncthreads();
  if (lane == 0) sh[wid] = v;
  __syncthreads();
  float r = 0.f;
#pragma unroll
  for (int i = 0; i < 8; ++i) r += sh[i];
  return r;
}
__device__ __forceinline__ float blockReduceMax256(float v, float* sh) {
  const int lane = threadIdx.x & 31, wid = threadIdx.x >> 5;
#pragma unroll
  for (int off = 16; off; off >>= 1) v = fmaxf(v, __shfl_xor(v, off, 32));
  __syncthreads();
  if (lane == 0) sh[wid] = v;
  __syncthreads();
  float r = -1e30f;
#pragma unroll
  for (int i = 0; i < 8; ++i) r = fmaxf(r, sh[i]);
  return r;
}

// ---------------- LayerNorm over D=768 --------------------------------------
__global__ __launch_bounds__(256) void ln_kernel(
    const float* __restrict__ x, const float* __restrict__ w,
    const float* __restrict__ b, float* __restrict__ out) {
  __shared__ float sh[8];
  const int row = blockIdx.x;
  const float* xr = x + (size_t)row * CD;
  float s = 0.f;
  for (int j = threadIdx.x; j < CD; j += 256) s += xr[j];
  const float mean = blockReduceSum256(s, sh) * (1.f / CD);
  float vs = 0.f;
  for (int j = threadIdx.x; j < CD; j += 256) {
    const float d = xr[j] - mean; vs += d * d;
  }
  const float var  = blockReduceSum256(vs, sh) * (1.f / CD);
  const float rstd = rsqrtf(var + 1e-5f);
  float* orow = out + (size_t)row * CD;
  for (int j = threadIdx.x; j < CD; j += 256)
    orow[j] = (xr[j] - mean) * rstd * w[j] + b[j];
}

// ---------------- Block-causal flash attention ------------------------------
// 1 wave per (b, h, 16-row q block). BS==16 == WMMA tile => mask exact at tile
// granularity (kt <= qt). LDS: scores 16x1024 (64KB) + K stage 2x16x64 (8KB,
// async-to-LDS double buffer) + V stage 2x32x64 (16KB, TDM double buffer).
__global__ __launch_bounds__(32) void attention_wmma(
    const float* __restrict__ qkv, float* __restrict__ o) {
  extern __shared__ float sc[];                 // scores
  float* kst = sc + 16 * CS;                    // 2 x 16 x 64
  float* vst = kst + 2 * 16 * CHD;              // 2 x 32 x 64
  const int qt = blockIdx.x;                    // 0..63
  const int h  = blockIdx.y;                    // 0..11
  const int b  = blockIdx.z;                    // 0..1
  const int lane = threadIdx.x;
  const int half = lane >> 4, r = lane & 15;
  const size_t rstride = 3 * CD;
  const float* base = qkv + (size_t)b * CS * rstride;
  const int nkt = qt + 1;
  const int nk  = nkt * 16;

  // Q tile A-operands for head-dim slices [0,32) and [32,64)
  const float* qrow = base + (size_t)(qt * 16 + r) * rstride + h * CHD;
  const v16bf qa0 = loadA_f32(qrow, 0,  half);
  const v16bf qa1 = loadA_f32(qrow, 32, half);
  const float scale = 0.125f;                   // 1/sqrt(64)

  // ---- async K-tile staging (double buffered) ------------------------------
  auto stage_k = [&](int kt, int buf) {
    const float* kb_base = base + (size_t)(kt * 16) * rstride + CD + h * CHD;
#pragma unroll
    for (int i = 0; i < 8; ++i) {
      const int idx = i * 32 + lane;            // 0..255 float4 transfers
      const int row = idx >> 4;                 // key row 0..15
      const int c4  = idx & 15;                 // float4 column
      const float* gp = kb_base + (size_t)row * rstride + c4 * 4;
      const unsigned ldsoff =
          (unsigned)(size_t)(kst + buf * 16 * CHD + row * CHD + c4 * 4);
      asm volatile("global_load_async_to_lds_b128 %0, %1, off"
                   :: "v"(ldsoff), "v"((unsigned long long)(size_t)gp)
                   : "memory");
    }
  };

  // ---- scores = scale * Q K^T into LDS -------------------------------------
  stage_k(0, 0);
  for (int kt = 0; kt < nkt; ++kt) {
    if (kt + 1 < nkt) {
      stage_k(kt + 1, (kt + 1) & 1);
      asm volatile("s_wait_asynccnt 0x8" ::: "memory");  // current tile done
    } else {
      asm volatile("s_wait_asynccnt 0x0" ::: "memory");
    }
    __syncthreads();
    const float* kr = kst + (kt & 1) * 16 * CHD + r * CHD;
    const v16bf kb0 = loadB_f32(kr, 0  + 16 * half);
    const v16bf kb1 = loadB_f32(kr, 32 + 16 * half);
    v8f c = {0.f, 0.f, 0.f, 0.f, 0.f, 0.f, 0.f, 0.f};
    c = wmma_bf16(qa0, kb0, c);
    c = wmma_bf16(qa1, kb1, c);
#pragma unroll
    for (int vr = 0; vr < 8; ++vr)
      sc[(vr + 8 * half) * CS + kt * 16 + r] = c[vr] * scale;
    __syncthreads();
  }
  __syncthreads();

  // ---- softmax per row (wave-cooperative) ----------------------------------
  for (int row = 0; row < 16; ++row) {
    float mx = -1e30f;
    for (int j = lane; j < nk; j += 32) mx = fmaxf(mx, sc[row * CS + j]);
#pragma unroll
    for (int off = 16; off; off >>= 1) mx = fmaxf(mx, __shfl_xor(mx, off, 32));
    float sum = 0.f;
    for (int j = lane; j < nk; j += 32) {
      const float e = __expf(sc[row * CS + j] - mx);
      sc[row * CS + j] = e;
      sum += e;
    }
#pragma unroll
    for (int off = 16; off; off >>= 1) sum += __shfl_xor(sum, off, 32);
    const float inv = 1.f / sum;
    for (int j = lane; j < nk; j += 32) sc[row * CS + j] *= inv;
  }
  // zero-pad score columns [nk, nk+16) when nkt is odd so the 32-wide P.V
  // k-steps read zeros instead of stale LDS.
  if (nkt & 1) {
    for (int i = lane; i < 256; i += 32)
      sc[(i >> 4) * CS + nk + (i & 15)] = 0.f;
  }
  __syncthreads();

  // ---- O(16x64) = P(16xnk) @ V(nk x 64) ------------------------------------
  v8f acc[4];
#pragma unroll
  for (int nt = 0; nt < 4; ++nt)
    acc[nt] = (v8f){0.f, 0.f, 0.f, 0.f, 0.f, 0.f, 0.f, 0.f};

  const float* vbase = base + 2 * CD + h * CHD;   // +token*rstride +dim

#if HAVE_TDM
  auto stage_v = [&](int kb, int buf) {
    tdm_load_2d((unsigned)(size_t)(vst + buf * 32 * CHD),
                vbase + (size_t)kb * rstride, 32u, (unsigned)CHD,
                (unsigned long long)rstride);
  };
  stage_v(0, 0);
#else
  auto stage_v = [&](int kb, int buf) {
#pragma unroll
    for (int i = 0; i < 16; ++i) {
      const int idx = i * 32 + lane;            // 0..511 float4 transfers
      const int row = idx >> 4;
      const int c4  = idx & 15;
      const float* gp = vbase + (size_t)(kb + row) * rstride + c4 * 4;
      const unsigned ldsoff =
          (unsigned)(size_t)(vst + buf * 32 * CHD + row * CHD + c4 * 4);
      asm volatile("global_load_async_to_lds_b128 %0, %1, off"
                   :: "v"(ldsoff), "v"((unsigned long long)(size_t)gp)
                   : "memory");
    }
  };
  stage_v(0, 0);
#endif

  for (int kb = 0; kb < nk; kb += 32) {
    const int cur = (kb >> 5) & 1;
    const bool more = (kb + 32) < nk;
    if (more) stage_v(kb + 32, cur ^ 1);
#if HAVE_TDM
    if (more) __builtin_amdgcn_s_wait_tensorcnt(1);
    else      __builtin_amdgcn_s_wait_tensorcnt(0);
#else
    if (more) asm volatile("s_wait_asynccnt 0x10" ::: "memory");
    else      asm volatile("s_wait_asynccnt 0x0"  ::: "memory");
#endif
    __syncthreads();

    // A operand from probability rows in LDS
    v16bf pa;
#pragma unroll
    for (int j = 0; j < 8; ++j)
      pa[j] = (__bf16)sc[r * CS + kb + 16 * half + j];
#pragma unroll
    for (int j = 0; j < 8; ++j)
      pa[8 + j] = (__bf16)sc[r * CS + kb + 16 * half + 8 + j];

    // B operands from the staged V tile in LDS
    const float* vt = vst + cur * 32 * CHD;
#pragma unroll
    for (int nt = 0; nt < 4; ++nt) {
      v16bf vb;
#pragma unroll
      for (int j = 0; j < 16; ++j)
        vb[j] = (__bf16)vt[(16 * half + j) * CHD + nt * 16 + r];
      acc[nt] = wmma_bf16(pa, vb, acc[nt]);
    }
    __syncthreads();
  }

#pragma unroll
  for (int nt = 0; nt < 4; ++nt)
#pragma unroll
    for (int vr = 0; vr < 8; ++vr) {
      const int rowg = qt * 16 + vr + 8 * half;
      o[(size_t)(b * CS + rowg) * CD + h * CHD + nt * 16 + r] = acc[nt][vr];
    }
}

// ---------------- GELU (exact, erf) ----------------------------------------
__global__ __launch_bounds__(256) void gelu_kernel(float* __restrict__ x,
                                                   long n) {
  for (long i = blockIdx.x * 256L + threadIdx.x; i < n; i += gridDim.x * 256L) {
    const float v = x[i];
    x[i] = 0.5f * v * (1.f + erff(v * 0.70710678118f));
  }
}

// ---------------- CE loss over vocab ---------------------------------------
__global__ void init_acc(float* acc) {
  if (threadIdx.x < 2) acc[threadIdx.x] = 0.f;
}

__global__ __launch_bounds__(256) void ce_kernel(
    const float* __restrict__ logits, const int* __restrict__ ids,
    const int* __restrict__ tgt, const float* __restrict__ noise,
    float* __restrict__ acc) {
  __shared__ float sh[8];
  const int row = blockIdx.x;                 // 0..N-1
  const float* lr = logits + (size_t)row * CV;
  float mx = -1e30f;
  for (int j = threadIdx.x; j < CV; j += 256) mx = fmaxf(mx, lr[j]);
  mx = blockReduceMax256(mx, sh);
  float s = 0.f;
  for (int j = threadIdx.x; j < CV; j += 256) s += __expf(lr[j] - mx);
  s = blockReduceSum256(s, sh);
  if (threadIdx.x == 0 && ids[row] == MASK_ID) {
    const float ce = -(lr[tgt[row]] - mx - __logf(s));
    atomicAdd(&acc[0], ce / noise[row / CS]);
    atomicAdd(&acc[1], 1.f);
  }
}

__global__ void finalize_loss(const float* __restrict__ acc,
                              float* __restrict__ out) {
  out[0] = acc[0] / (acc[1] + 1e-8f);
}

// ---------------- Host orchestration ---------------------------------------
static inline void launch_gemm(const float* A, const float* W,
                               const float* bias, const float* resid, float* C,
                               int M, int N, int K, hipStream_t s) {
  dim3 grid(N / 256, M / 64);
  if (resid)
    gemm_bf16_wmma<true><<<grid, 256, 0, s>>>(A, W, bias, resid, C, M, N, K);
  else
    gemm_bf16_wmma<false><<<grid, 256, 0, s>>>(A, W, bias, nullptr, C, M, N, K);
}

extern "C" void kernel_launch(void* const* d_in, const int* in_sizes, int n_in,
                              void* d_out, int out_size, void* d_ws,
                              size_t ws_size, hipStream_t stream) {
  const int*   input_ids  = (const int*)d_in[0];
  const int*   target_ids = (const int*)d_in[1];
  const float* noise      = (const float*)d_in[2];
  const float* emb        = (const float*)d_in[3];
  const float* in_w       = (const float*)d_in[4];
  const float* in_b       = (const float*)d_in[5];
  const float* aow        = (const float*)d_in[6];
  const float* aob        = (const float*)d_in[7];
  const float* ln1w       = (const float*)d_in[8];
  const float* ln1b       = (const float*)d_in[9];
  const float* ln2w       = (const float*)d_in[10];
  const float* ln2b       = (const float*)d_in[11];
  const float* w1         = (const float*)d_in[12];
  const float* b1         = (const float*)d_in[13];
  const float* w2         = (const float*)d_in[14];
  const float* b2         = (const float*)d_in[15];
  const float* projw      = (const float*)d_in[16];
  const float* projb      = (const float*)d_in[17];

  float* out    = (float*)d_out;
  float* logits = out + 1;                    // [N, V]

  float* ws   = (float*)d_ws;
  float* x    = ws;                           // [N, D]
  float* h    = x + (size_t)CN * CD;          // [N, D]
  float* qkvb = h + (size_t)CN * CD;          // [N, 3D]
  float* ao   = qkvb + (size_t)CN * 3 * CD;   // [N, D]
  float* ffn  = ao + (size_t)CN * CD;         // [N, DF]
  float* acc  = ffn + (size_t)CN * CDF;       // [2]

  // Embedding + rotary
  embed_rotary<<<CN, 256, 0, stream>>>(input_ids, emb, x);

  const size_t attn_lds =
      (16 * CS + 2 * 16 * CHD + 2 * 32 * CHD) * sizeof(float);  // 88KB

  for (int l = 0; l < CL; ++l) {
    // h = LN1(x)
    ln_kernel<<<CN, 256, 0, stream>>>(x, ln1w + l * CD, ln1b + l * CD, h);
    // qkv = h @ in_w[l]^T + in_b[l]
    launch_gemm(h, in_w + (size_t)l * 3 * CD * CD, in_b + (size_t)l * 3 * CD,
                nullptr, qkvb, CN, 3 * CD, CD, stream);
    // attention (block-causal, WMMA + async K staging + TDM V staging)
    attention_wmma<<<dim3(CS / 16, CH, CB), 32, attn_lds, stream>>>(qkvb, ao);
    // x = x + ao @ attn_out_w[l]^T + attn_out_b[l]
    launch_gemm(ao, aow + (size_t)l * CD * CD, aob + (size_t)l * CD, x, x, CN,
                CD, CD, stream);
    // h = LN2(x)
    ln_kernel<<<CN, 256, 0, stream>>>(x, ln2w + l * CD, ln2b + l * CD, h);
    // ffn = gelu(h @ w1^T + b1)
    launch_gemm(h, w1 + (size_t)l * CDF * CD, b1 + (size_t)l * CDF, nullptr,
                ffn, CN, CDF, CD, stream);
    gelu_kernel<<<1024, 256, 0, stream>>>(ffn, (long)CN * CDF);
    // x = x + ffn @ w2^T + b2
    launch_gemm(ffn, w2 + (size_t)l * CD * CDF, b2 + (size_t)l * CD, x, x, CN,
                CD, CDF, stream);
  }

  // logits = x @ proj_w^T + proj_b  (written straight into d_out+1)
  launch_gemm(x, projw, projb, nullptr, logits, CN, CV, CD, stream);

  // masked, noise-weighted CE
  init_acc<<<1, 32, 0, stream>>>(acc);
  ce_kernel<<<CN, 256, 0, stream>>>(logits, input_ids, target_ids, noise, acc);
  finalize_loss<<<1, 1, 0, stream>>>(acc, out);
}
